// SE_GATNoEdgeAttrsSelectiveMask_59365037966023
// MI455X (gfx1250) — matrix operand order repaired and checked by reading, not compiled, not run on hardware
//
#include <hip/hip_runtime.h>
#include <hip/hip_bf16.h>

#define NNODES  50000
#define NEDGES  400000
#define FIN     128
#define HEADS   4
#define C1V     64
#define C2V     16
#define OUTD    2
#define NGRAPHS 64

typedef __attribute__((ext_vector_type(16))) _Float16 v16h;
typedef __attribute__((ext_vector_type(8)))  _Float16 v8h;
typedef __attribute__((ext_vector_type(8)))  float    v8f;

__device__ __forceinline__ float leaky(float z) { return z > 0.f ? z : 0.2f * z; }

// order-preserving float<->uint for atomicMax on signed floats
__device__ __forceinline__ unsigned ford(float f) {
    unsigned u = __float_as_uint(f);
    return (u & 0x80000000u) ? ~u : (u | 0x80000000u);
}
__device__ __forceinline__ float deord(unsigned o) {
    unsigned u = (o & 0x80000000u) ? (o & 0x7fffffffu) : ~o;
    return __uint_as_float(u);
}

// ---------------------------------------------------------------------------
// Wave-level 16x16 output tile GEMM, f16 inputs, f32 accumulate, K multiple
// of 32.  Fragment layouts per CDNA5 ISA 7.12.2 (wave32):
//   A 16x32: lane L -> row M=L%16; halves [k0+8*(L/16), +8) and +16 halves.
//   B 32x16: lane L -> col N=L%16; rows [k0+8*(L/16), +8) and +16 rows.
//   C/D:     VGPR v, lane L -> M = v + 8*(L/16), N = L%16.
// ---------------------------------------------------------------------------
template <int K>
__device__ __forceinline__ void wave_gemm_tile(const _Float16* __restrict__ A, int lda,
                                               const _Float16* __restrict__ B, int ldb,
                                               float* __restrict__ D, int ldd) {
    const int lane = threadIdx.x & 31;
    const int hl   = lane >> 4;   // K-half selector
    const int r    = lane & 15;   // row within tile
    v8f acc = {};
#pragma unroll
    for (int k0 = 0; k0 < K; k0 += 32) {
        const _Float16* pa = A + (size_t)r * lda + k0 + hl * 8;
        v8h alo = *(const v8h*)(pa);
        v8h ahi = *(const v8h*)(pa + 16);
        v16h a;
#pragma unroll
        for (int i = 0; i < 8; ++i) { a[i] = alo[i]; a[i + 8] = ahi[i]; }
        v16h b;
#pragma unroll
        for (int i = 0; i < 8; ++i) {
            b[i]     = B[(size_t)(k0 + hl * 8 + i) * ldb + r];
            b[i + 8] = B[(size_t)(k0 + 16 + hl * 8 + i) * ldb + r];
        }
        acc = __builtin_amdgcn_wmma_f32_16x16x32_f16(false, a, false, b,
                                                     (short)0, acc, false, false);
    }
#pragma unroll
    for (int v = 0; v < 8; ++v)
        D[(size_t)(hl * 8 + v) * ldd + r] = acc[v];
}

// GEMM1: [50000,128] x [128,256] -> [50000,256].  16 waves/block cover N=256.
__global__ void k_gemm1(const _Float16* __restrict__ xh, const _Float16* __restrict__ Wh,
                        float* __restrict__ h1) {
    const int m0 = blockIdx.x * 16;
    const int n0 = (threadIdx.x >> 5) * 16;
    wave_gemm_tile<FIN>(xh + (size_t)m0 * FIN, FIN, Wh + n0, HEADS * C1V,
                        h1 + (size_t)m0 * (HEADS * C1V) + n0, HEADS * C1V);
}

// GEMM2: [50000,256] x [256,64] -> [50000,64].  4 waves/block cover N=64.
__global__ void k_gemm2(const _Float16* __restrict__ hh, const _Float16* __restrict__ Wh,
                        float* __restrict__ h2) {
    const int m0 = blockIdx.x * 16;
    const int n0 = (threadIdx.x >> 5) * 16;
    wave_gemm_tile<HEADS * C1V>(hh + (size_t)m0 * (HEADS * C1V), HEADS * C1V, Wh + n0,
                                HEADS * C2V, h2 + (size_t)m0 * (HEADS * C2V) + n0,
                                HEADS * C2V);
}

__global__ void k_f32_to_f16(const float* __restrict__ s, _Float16* __restrict__ d, int n) {
    int i = blockIdx.x * blockDim.x + threadIdx.x;
    if (i < n) d[i] = (_Float16)s[i];
}

__global__ void k_zero(float* __restrict__ p, int n) {
    int i = blockIdx.x * blockDim.x + threadIdx.x;
    if (i < n) p[i] = 0.f;
}

// alpha_{s,d}[n,h] = sum_c h[n,h,c] * a_{s,d}[h,c]
template <int C>
__global__ void k_alpha(const float* __restrict__ h, const float* __restrict__ a_s,
                        const float* __restrict__ a_d, float* __restrict__ as,
                        float* __restrict__ ad) {
    int i = blockIdx.x * blockDim.x + threadIdx.x;
    if (i >= NNODES * HEADS) return;
    int n = i >> 2, hh = i & 3;
    const float* hp = h + (size_t)n * HEADS * C + hh * C;
    float ss = 0.f, dd = 0.f;
#pragma unroll
    for (int c = 0; c < C; ++c) { float v = hp[c]; ss += v * a_s[hh * C + c]; dd += v * a_d[hh * C + c]; }
    as[i] = ss; ad[i] = dd;
}

// seed running max with the self-loop edge value
__global__ void k_init_max(const float* __restrict__ as, const float* __restrict__ ad,
                           unsigned* __restrict__ mord) {
    int i = blockIdx.x * blockDim.x + threadIdx.x;
    if (i >= NNODES * HEADS) return;
    mord[i] = ford(leaky(as[i] + ad[i]));
}

__global__ void k_edge_max(const int* __restrict__ src, const int* __restrict__ dst,
                           const float* __restrict__ as, const float* __restrict__ ad,
                           unsigned* __restrict__ mord) {
    int e = blockIdx.x * blockDim.x + threadIdx.x;
    if (e >= NEDGES) return;
    int s = src[e], d = dst[e];
#pragma unroll
    for (int hh = 0; hh < HEADS; ++hh) {
        float z = leaky(as[s * HEADS + hh] + ad[d * HEADS + hh]);
        atomicMax(&mord[d * HEADS + hh], ford(z));
    }
}

// decode max in place; init denominator with self-loop contribution
__global__ void k_fin_max(unsigned* __restrict__ mord, const float* __restrict__ as,
                          const float* __restrict__ ad, float* __restrict__ den) {
    int i = blockIdx.x * blockDim.x + threadIdx.x;
    if (i >= NNODES * HEADS) return;
    float m  = deord(mord[i]);
    float es = leaky(as[i] + ad[i]);
    ((float*)mord)[i] = m;
    den[i] = expf(es - m);
}

__global__ void k_edge_den(const int* __restrict__ src, const int* __restrict__ dst,
                           const float* __restrict__ as, const float* __restrict__ ad,
                           const float* __restrict__ m, float* __restrict__ den) {
    int t = blockIdx.x * blockDim.x + threadIdx.x;
    if (t >= NEDGES * HEADS) return;
    int e = t >> 2, hh = t & 3;
    int s = src[e], d = dst[e];
    int nh = d * HEADS + hh;
    float z = leaky(as[s * HEADS + hh] + ad[nh]);
    atomicAdd(&den[nh], expf(z - m[nh]));
}

// out = alpha_self * h   (covers self-loop edge, initializes accumulator)
template <int C>
__global__ void k_agg_init(const float* __restrict__ h, const float* __restrict__ as,
                           const float* __restrict__ ad, const float* __restrict__ m,
                           const float* __restrict__ den, float* __restrict__ out) {
    int i = blockIdx.x * blockDim.x + threadIdx.x;
    if (i >= NNODES * HEADS * C) return;
    int nh = i / C;  // n*HEADS + hh
    float a = expf(leaky(as[nh] + ad[nh]) - m[nh]) / (den[nh] + 1e-16f);
    out[i] = a * h[i];
}

// one thread per (edge, channel): lane-coalesced gather of h[src] row and
// lane-coalesced f32 atomic scatter into out[dst] row (both L2-resident).
template <int C>
__global__ void k_edge_agg(const int* __restrict__ src, const int* __restrict__ dst,
                           const float* __restrict__ as, const float* __restrict__ ad,
                           const float* __restrict__ m, const float* __restrict__ den,
                           const float* __restrict__ h, float* __restrict__ out) {
    int t = blockIdx.x * blockDim.x + threadIdx.x;
    if (t >= NEDGES * HEADS * C) return;
    int e   = t / (HEADS * C);
    int rem = t & (HEADS * C - 1);   // hh*C + c   (HEADS*C is a power of two)
    int hh  = rem / C;
    int s = src[e], d = dst[e];
    int nh = d * HEADS + hh;
    float w = expf(leaky(as[s * HEADS + hh] + ad[nh]) - m[nh]) / (den[nh] + 1e-16f);
    atomicAdd(&out[(size_t)d * HEADS * C + rem], w * h[(size_t)s * HEADS * C + rem]);
}

// o = relu(o + bias); optionally emit f16 copy for the next WMMA GEMM
__global__ void k_post(float* __restrict__ o, const float* __restrict__ b,
                       _Float16* __restrict__ h16, int total, int ld) {
    int i = blockIdx.x * blockDim.x + threadIdx.x;
    if (i >= total) return;
    float v = o[i] + b[i % ld];
    v = v > 0.f ? v : 0.f;
    o[i] = v;
    if (h16) h16[i] = (_Float16)v;
}

// one thread per (node, channel): coalesced atomics into 16KB pool buffer
__global__ void k_pool(const float* __restrict__ h, const int* __restrict__ batch,
                       float* __restrict__ sums, float* __restrict__ cnt) {
    int t = blockIdx.x * blockDim.x + threadIdx.x;
    if (t >= NNODES * (HEADS * C2V)) return;
    int n = t / (HEADS * C2V);
    int j = t & (HEADS * C2V - 1);
    int b = batch[n];
    atomicAdd(&sums[(size_t)b * (HEADS * C2V) + j], h[t]);
    if (j == 0) atomicAdd(&cnt[b], 1.0f);
}

__global__ void k_fc(const float* __restrict__ sums, const float* __restrict__ cnt,
                     const float* __restrict__ Wfc, const float* __restrict__ bfc,
                     float* __restrict__ out) {
    int i = threadIdx.x;
    if (i >= NGRAPHS * OUTD) return;
    int g = i >> 1, o = i & 1;
    float c = cnt[g]; c = c > 1.f ? c : 1.f;
    float acc = bfc[o];
#pragma unroll
    for (int j = 0; j < HEADS * C2V; ++j) acc += (sums[g * (HEADS * C2V) + j] / c) * Wfc[j * OUTD + o];
    out[i] = acc;
}

extern "C" void kernel_launch(void* const* d_in, const int* in_sizes, int n_in,
                              void* d_out, int out_size, void* d_ws, size_t ws_size,
                              hipStream_t stream) {
    const float* x      = (const float*)d_in[0];
    const int*   ei     = (const int*)d_in[1];
    const int*   batch  = (const int*)d_in[2];
    const float* W1     = (const float*)d_in[3];
    const float* a_src1 = (const float*)d_in[4];
    const float* a_dst1 = (const float*)d_in[5];
    const float* b1     = (const float*)d_in[6];
    const float* W2     = (const float*)d_in[7];
    const float* a_src2 = (const float*)d_in[8];
    const float* a_dst2 = (const float*)d_in[9];
    const float* b2     = (const float*)d_in[10];
    const float* Wfc    = (const float*)d_in[11];
    const float* bfc    = (const float*)d_in[12];
    float* out = (float*)d_out;

    const int* src1 = ei + 17 * NEDGES;
    const int* dst1 = ei + 26 * NEDGES;
    const int* src2 = ei + 15 * NEDGES;
    const int* dst2 = ei + 16 * NEDGES;

    // workspace carve-up (256B aligned)
    char* base = (char*)d_ws;
    size_t off = 0;
    auto carve = [&](size_t bytes) -> char* {
        char* p = base + off;
        off = (off + bytes + 255) & ~(size_t)255;
        return p;
    };
    _Float16* xh   = (_Float16*)carve((size_t)NNODES * FIN * 2);
    _Float16* W1h  = (_Float16*)carve((size_t)FIN * HEADS * C1V * 2);
    _Float16* W2h  = (_Float16*)carve((size_t)(HEADS * C1V) * (HEADS * C2V) * 2);
    float*    h1   = (float*)carve((size_t)NNODES * HEADS * C1V * 4);
    float*    out1 = (float*)carve((size_t)NNODES * HEADS * C1V * 4);
    _Float16* h1h  = (_Float16*)carve((size_t)NNODES * HEADS * C1V * 2);
    float*    h2   = (float*)carve((size_t)NNODES * HEADS * C2V * 4);
    float*    out2 = (float*)carve((size_t)NNODES * HEADS * C2V * 4);
    float*    as1  = (float*)carve((size_t)NNODES * HEADS * 4);
    float*    ad1  = (float*)carve((size_t)NNODES * HEADS * 4);
    float*    m1   = (float*)carve((size_t)NNODES * HEADS * 4);
    float*    den1 = (float*)carve((size_t)NNODES * HEADS * 4);
    float*    as2  = (float*)carve((size_t)NNODES * HEADS * 4);
    float*    ad2  = (float*)carve((size_t)NNODES * HEADS * 4);
    float*    m2   = (float*)carve((size_t)NNODES * HEADS * 4);
    float*    den2 = (float*)carve((size_t)NNODES * HEADS * 4);
    float*    sums = (float*)carve((size_t)(NGRAPHS * HEADS * C2V + NGRAPHS) * 4);
    float*    cnt  = sums + NGRAPHS * HEADS * C2V;

    auto G = [](int n, int b) { return (n + b - 1) / b; };
    const int NH = NNODES * HEADS;

    // ---- precision prep: f32 -> f16 for WMMA operands
    k_f32_to_f16<<<G(NNODES * FIN, 256), 256, 0, stream>>>(x, xh, NNODES * FIN);
    k_f32_to_f16<<<G(FIN * HEADS * C1V, 256), 256, 0, stream>>>(W1, W1h, FIN * HEADS * C1V);
    k_f32_to_f16<<<G((HEADS * C1V) * (HEADS * C2V), 256), 256, 0, stream>>>(
        W2, W2h, (HEADS * C1V) * (HEADS * C2V));

    // ---- layer 1: GEMM (WMMA) + attention softmax + aggregate
    k_gemm1<<<NNODES / 16, 512, 0, stream>>>(xh, W1h, h1);
    k_alpha<C1V><<<G(NH, 256), 256, 0, stream>>>(h1, a_src1, a_dst1, as1, ad1);
    k_init_max<<<G(NH, 256), 256, 0, stream>>>(as1, ad1, (unsigned*)m1);
    k_edge_max<<<G(NEDGES, 256), 256, 0, stream>>>(src1, dst1, as1, ad1, (unsigned*)m1);
    k_fin_max<<<G(NH, 256), 256, 0, stream>>>((unsigned*)m1, as1, ad1, den1);
    k_edge_den<<<G(NEDGES * HEADS, 256), 256, 0, stream>>>(src1, dst1, as1, ad1, m1, den1);
    k_agg_init<C1V><<<G(NNODES * HEADS * C1V, 256), 256, 0, stream>>>(h1, as1, ad1, m1, den1, out1);
    k_edge_agg<C1V><<<G(NEDGES * HEADS * C1V, 256), 256, 0, stream>>>(src1, dst1, as1, ad1,
                                                                      m1, den1, h1, out1);
    k_post<<<G(NNODES * HEADS * C1V, 256), 256, 0, stream>>>(out1, b1, h1h,
                                                             NNODES * HEADS * C1V, HEADS * C1V);

    // ---- layer 2
    k_gemm2<<<NNODES / 16, 128, 0, stream>>>(h1h, W2h, h2);
    k_alpha<C2V><<<G(NH, 256), 256, 0, stream>>>(h2, a_src2, a_dst2, as2, ad2);
    k_init_max<<<G(NH, 256), 256, 0, stream>>>(as2, ad2, (unsigned*)m2);
    k_edge_max<<<G(NEDGES, 256), 256, 0, stream>>>(src2, dst2, as2, ad2, (unsigned*)m2);
    k_fin_max<<<G(NH, 256), 256, 0, stream>>>((unsigned*)m2, as2, ad2, den2);
    k_edge_den<<<G(NEDGES * HEADS, 256), 256, 0, stream>>>(src2, dst2, as2, ad2, m2, den2);
    k_agg_init<C2V><<<G(NNODES * HEADS * C2V, 256), 256, 0, stream>>>(h2, as2, ad2, m2, den2, out2);
    k_edge_agg<C2V><<<G(NEDGES * HEADS * C2V, 256), 256, 0, stream>>>(src2, dst2, as2, ad2,
                                                                      m2, den2, h2, out2);
    k_post<<<G(NNODES * HEADS * C2V, 256), 256, 0, stream>>>(out2, b2, (_Float16*)nullptr,
                                                             NNODES * HEADS * C2V, HEADS * C2V);

    // ---- global mean pool + FC
    k_zero<<<G(NGRAPHS * HEADS * C2V + NGRAPHS, 256), 256, 0, stream>>>(
        sums, NGRAPHS * HEADS * C2V + NGRAPHS);
    k_pool<<<G(NNODES * HEADS * C2V, 256), 256, 0, stream>>>(out2, batch, sums, cnt);
    k_fc<<<1, 128, 0, stream>>>(sums, cnt, Wfc, bfc, out);
}
